// Model_91242285236312
// MI455X (gfx1250) — compile-verified, matrix operand order checked by reading, not run on hardware
//
#include <hip/hip_runtime.h>
#include <hip/hip_bf16.h>

// ---------------------------------------------------------------------------
// DIF-transformer (B=256,S=200,H=64,NH=2,E=768,NB=2) for MI455X / gfx1250.
// bf16 WMMA (v_wmma_f32_16x16x32_bf16) for all GEMMs; gather fused with the
// E=768 feature GEMM; attention folds item+pos+feat score terms into one
// 96-channel Q/K concat GEMM per head. Weights are pre-converted once to
// transposed bf16 so every GEMM kernel stages them with a single Tensor Data
// Mover DMA (tensor_load_to_lds + s_wait_tensorcnt); attention K/V slabs are
// TDM'd likewise.
// ---------------------------------------------------------------------------

typedef __bf16 bf16_t;
typedef __attribute__((ext_vector_type(16))) __bf16 v16bf;
typedef __attribute__((ext_vector_type(8)))  __bf16 bf16x8;
typedef __attribute__((ext_vector_type(8)))  float  v8f;
typedef __attribute__((ext_vector_type(4)))  float  f32x4;
typedef unsigned int u32x4 __attribute__((ext_vector_type(4)));
typedef int          i32x4 __attribute__((ext_vector_type(4)));
typedef int          i32x8 __attribute__((ext_vector_type(8)));

#define BDIM   256
#define SLEN   200
#define NHEAD  2
#define HDIM   64
#define EDIM   768
#define BS     (BDIM * SLEN)      // 51200 rows
#define SP     208                // S padded to 13*16
#define SPP    224                // probs row padded to 7*32
#define NQT    13

#if defined(__has_builtin)
#  if __has_builtin(__builtin_amdgcn_tensor_load_to_lds)
#    define HAVE_TDM 1
#  endif
#endif
#ifndef HAVE_TDM
#  define HAVE_TDM 0
#endif

// Attention LDS carve (dynamic shared, so offsets start at 0 for TDM lds_addr)
#define LDS_K_OFF   0                         // 200*96 bf16 = 38400 B
#define LDS_V_OFF   38400                     // 32*208 bf16 = 13312 B
#define LDS_SC_OFF  (38400 + 13312)           // 16*208 f32  = 13312 B
#define LDS_SP_OFF  (38400 + 13312 + 13312)   // 16*224 bf16 =  7168 B
#define ATTN_LDS_BYTES (38400 + 13312 + 13312 + 7168)

// feat kernel LDS carve
#define FEAT_W_BYTES   (64 * EDIM * 2)        // 98304 B transposed bf16 featW
#define FEAT_LDS_BYTES (FEAT_W_BYTES + 64 * 64 * 4)

__device__ inline v8f wmma_bf16(v16bf a, v16bf b, v8f c) {
    return __builtin_amdgcn_wmma_f32_16x16x32_bf16(
        false, a, false, b, (short)0, c, false, false);
}

// A/B fragment for 16-bit WMMA from a row-major row pointer.
// elements [0..7]  = row[kBase + half*8 + i]
// elements [8..15] = row[kBase + 16 + half*8 + i]
__device__ inline v16bf frag_row(const bf16_t* row, int kBase, int half) {
    bf16x8 lo = *(const bf16x8*)(row + kBase + half * 8);
    bf16x8 hi = *(const bf16x8*)(row + kBase + 16 + half * 8);
    v16bf a;
#pragma unroll
    for (int i = 0; i < 8; ++i) { a[i] = lo[i]; a[i + 8] = hi[i]; }
    return a;
}

__device__ inline v16bf frag_zero() {
    v16bf a;
#pragma unroll
    for (int i = 0; i < 16; ++i) a[i] = (bf16_t)0.0f;
    return a;
}

#if HAVE_TDM
// 1-D TDM copy: nelem bf16 elements global -> LDS (descriptor per ISA 8.3/8.4).
__device__ inline void tdm_load_1d(unsigned long long gaddr, unsigned ldsOff,
                                   unsigned nelem) {
    u32x4 g0;
    g0[0] = 1u;                                        // count=1 (valid), user
    g0[1] = ldsOff;                                    // lds_addr (bytes)
    g0[2] = (unsigned)(gaddr & 0xffffffffu);           // global_addr[31:0]
    g0[3] = (unsigned)((gaddr >> 32) & 0x01ffffffu)    // global_addr[56:32]
            | (2u << 30);                              // type = 2 ("image")
    i32x8 g1;
    g1[0] = (1 << 16);                                 // data_size=1 (2B), mask=0
    g1[1] = (int)((nelem & 0xffffu) << 16);            // tensor_dim0[15:0]
    g1[2] = (int)((nelem >> 16) & 0xffffu) | (1 << 16);// dim0 hi | tensor_dim1=1
    g1[3] = (int)((nelem & 0xffffu) << 16);            // tile_dim0 (<=65535)
    g1[4] = 1;                                         // tile_dim1=1, tile_dim2=0
    g1[5] = (int)nelem;                                // tensor_dim0_stride lo
    g1[6] = 0;                                         // stride hi / dim1 stride
    g1[7] = 0;
    i32x4 z4 = {0, 0, 0, 0};
#if __clang_major__ >= 23
    i32x8 z8 = {0, 0, 0, 0, 0, 0, 0, 0};
    __builtin_amdgcn_tensor_load_to_lds(g0, g1, z4, z4, z8, 0);
#else
    __builtin_amdgcn_tensor_load_to_lds(g0, g1, z4, z4, 0);
#endif
}
#endif

// ---------------------------------------------------------------------------
// One-shot weight prep: all 20 [64x64] mats + featW -> transposed bf16.
// WtAll[(j*2+i)][n*64+k] = W_j[i][k][n];  featWt[n*768+k] = featW[k][n]
// ---------------------------------------------------------------------------
__global__ void prep_weights(const float* __restrict__ w0, const float* __restrict__ w1,
                             const float* __restrict__ w2, const float* __restrict__ w3,
                             const float* __restrict__ w4, const float* __restrict__ w5,
                             const float* __restrict__ w6, const float* __restrict__ w7,
                             const float* __restrict__ w8, const float* __restrict__ w9,
                             const float* __restrict__ featW,
                             bf16_t* __restrict__ WtAll, bf16_t* __restrict__ featWt) {
    int idx = blockIdx.x * 256 + threadIdx.x;
    const int NW = 20 * 4096;
    if (idx < NW) {
        int mat = idx >> 12, e = idx & 4095;
        int j = mat >> 1, i = mat & 1;
        int k = e >> 6, n = e & 63;
        const float* srcs[10] = {w0, w1, w2, w3, w4, w5, w6, w7, w8, w9};
        WtAll[(long)mat * 4096 + n * 64 + k] = (bf16_t)srcs[j][i * 4096 + e];
    } else if (idx < NW + 64 * EDIM) {
        int e = idx - NW;                 // e = k*64+n over featW
        int k = e >> 6, n = e & 63;
        featWt[(long)n * EDIM + k] = (bf16_t)featW[e];
    }
}

// ---------------------------------------------------------------------------
// init: q = item_emb[targetSeq] * keep  (f32 + bf16),  pe table -> bf16
// ---------------------------------------------------------------------------
__global__ void init_kernel(const int* __restrict__ ts,
                            const float* __restrict__ item_emb,
                            const float* __restrict__ pe_tab,
                            float* __restrict__ qF, bf16_t* __restrict__ qB,
                            bf16_t* __restrict__ peB) {
    int i = blockIdx.x * 256 + threadIdx.x;
    if (i < BS * 64) {
        int m = i >> 6, c = i & 63;
        int t = ts[m];
        float v = item_emb[(long)t * 64 + c] * ((t != 0) ? 1.0f : 0.0f);
        qF[i] = v; qB[i] = (bf16_t)v;
    }
    if (i < SLEN * 64) peB[i] = (bf16_t)pe_tab[i];
}

// ---------------------------------------------------------------------------
// feat = LN((rev[t]+meta[t]) @ featW + featb)   [BS x 768] x [768 x 64]
// gather fused with GEMM; whole transposed featW TDM'd to LDS once.
// ---------------------------------------------------------------------------
__global__ __launch_bounds__(128) void feat_kernel(
    const int* __restrict__ ts, const float* __restrict__ rev,
    const float* __restrict__ meta, const bf16_t* __restrict__ featWt,
    const float* __restrict__ featb, const float* __restrict__ g,
    const float* __restrict__ bta, bf16_t* __restrict__ featOut) {
    extern __shared__ char smem[];
    bf16_t* ldsW   = (bf16_t*)smem;                   // [64][768] Wt
    float*  ldsOut = (float*)(smem + FEAT_W_BYTES);   // [64][64]

    int tid = threadIdx.x;
    int wave = tid >> 5, lane = tid & 31, lm = lane & 15, half = lane >> 4;
    int rowBase = blockIdx.x * 64;
    int aRow = rowBase + wave * 16 + lm;
    int t = ts[aRow];
    const float* r0 = rev  + (long)t * EDIM;
    const float* m0 = meta + (long)t * EDIM;

#if HAVE_TDM
    if (wave == 0) {
        tdm_load_1d((unsigned long long)(uintptr_t)featWt, 0, 64 * EDIM);
        __builtin_amdgcn_s_wait_tensorcnt(0);
    }
#else
    for (int i = tid; i < 64 * EDIM / 8; i += 128)
        ((bf16x8*)ldsW)[i] = ((const bf16x8*)featWt)[i];
#endif
    __syncthreads();

    v8f acc[4] = {};
    for (int kc = 0; kc < EDIM / 32; ++kc) {
        int kBase = kc * 32;
        if (kc + 1 < EDIM / 32) {         // stream-ahead for the gather rows
            __builtin_prefetch(r0 + kBase + 32 + half * 8, 0, 3);
            __builtin_prefetch(m0 + kBase + 32 + half * 8, 0, 3);
        }
        // gather A fragment (sum of two tables), fp32 -> bf16
        v16bf a;
#pragma unroll
        for (int g2 = 0; g2 < 2; ++g2) {
            int c0 = kBase + g2 * 16 + half * 8;
            f32x4 xa = *(const f32x4*)(r0 + c0);
            f32x4 xb = *(const f32x4*)(r0 + c0 + 4);
            f32x4 ya = *(const f32x4*)(m0 + c0);
            f32x4 yb = *(const f32x4*)(m0 + c0 + 4);
#pragma unroll
            for (int i = 0; i < 4; ++i) {
                a[g2 * 8 + i]     = (bf16_t)(xa[i] + ya[i]);
                a[g2 * 8 + 4 + i] = (bf16_t)(xb[i] + yb[i]);
            }
        }
#pragma unroll
        for (int nt = 0; nt < 4; ++nt) {
            v16bf bf = frag_row(ldsW + (long)(nt * 16 + lm) * EDIM, kBase, half);
            acc[nt] = wmma_bf16(a, bf, acc[nt]);
        }
    }
    __syncthreads();
#pragma unroll
    for (int nt = 0; nt < 4; ++nt)
#pragma unroll
        for (int r = 0; r < 8; ++r) {
            int col = nt * 16 + lm;
            ldsOut[(wave * 16 + r + half * 8) * 64 + col] = acc[nt][r] + featb[col];
        }
    __syncthreads();
    if (tid < 64) {     // per-row LayerNorm (eps 1e-5) -> bf16 out
        const float* x = ldsOut + tid * 64;
        float s = 0.f;
        for (int c = 0; c < 64; ++c) s += x[c];
        float mean = s * (1.0f / 64.0f), v = 0.f;
        for (int c = 0; c < 64; ++c) { float d = x[c] - mean; v += d * d; }
        float inv = rsqrtf(v * (1.0f / 64.0f) + 1e-5f);
        bf16_t* out = featOut + (long)(rowBase + tid) * 64;
        for (int c = 0; c < 64; ++c)
            out[c] = (bf16_t)((x[c] - mean) * inv * g[c] + bta[c]);
    }
}

// ---------------------------------------------------------------------------
// Generic C[Mx64] = A[Mx64](bf16) @ Wt[64x64](bf16, pre-transposed) + bias.
// Wt slab DMA'd to LDS by the TDM. mode: 0 plain, 1 cat, 2 V-transposed.
// ---------------------------------------------------------------------------
#define GM_PLAIN 0
#define GM_CAT   1
#define GM_VT    2

__global__ __launch_bounds__(128) void gemm64_kernel(
    const bf16_t* __restrict__ A, const bf16_t* __restrict__ Wt,
    const float* __restrict__ bias, const float* __restrict__ resid,
    const int* __restrict__ keepIdx, float* __restrict__ outF,
    bf16_t* __restrict__ outB, int M, int relu, int mode, int catOff) {
    extern __shared__ char smem[];
    bf16_t* ldsWt = (bf16_t*)smem;        // [64][64] Wt
    int tid = threadIdx.x;
    int wave = tid >> 5, lane = tid & 31, lm = lane & 15, half = lane >> 4;
#if HAVE_TDM
    if (wave == 0) {
        tdm_load_1d((unsigned long long)(uintptr_t)Wt, 0, 64 * 64);
        __builtin_amdgcn_s_wait_tensorcnt(0);
    }
#else
    for (int i = tid; i < 64 * 64 / 8; i += 128)
        ((bf16x8*)ldsWt)[i] = ((const bf16x8*)Wt)[i];
#endif
    __syncthreads();

    int rowBase = blockIdx.x * 64 + wave * 16;
    int aRow = rowBase + lm;
    v16bf a0, a1;
    if (aRow < M) {
        a0 = frag_row(A + (long)aRow * 64, 0, half);
        a1 = frag_row(A + (long)aRow * 64, 32, half);
    } else { a0 = frag_zero(); a1 = frag_zero(); }

    v8f acc[4] = {};
#pragma unroll
    for (int nt = 0; nt < 4; ++nt) {
        v16bf b0 = frag_row(ldsWt + (nt * 16 + lm) * 64, 0, half);
        acc[nt] = wmma_bf16(a0, b0, acc[nt]);
        v16bf b1 = frag_row(ldsWt + (nt * 16 + lm) * 64, 32, half);
        acc[nt] = wmma_bf16(a1, b1, acc[nt]);
    }

#pragma unroll
    for (int nt = 0; nt < 4; ++nt)
#pragma unroll
        for (int r = 0; r < 8; ++r) {
            int m = rowBase + r + half * 8;
            if (m >= M) continue;
            int col = nt * 16 + lm;
            float v = acc[nt][r] + bias[col];
            if (resid)   v += resid[(long)m * 64 + col];
            if (relu)    v = fmaxf(v, 0.0f);
            if (keepIdx) v *= (keepIdx[m] != 0) ? 1.0f : 0.0f;
            if (mode == GM_PLAIN) {
                if (outF) outF[(long)m * 64 + col] = v;
                if (outB) outB[(long)m * 64 + col] = (bf16_t)v;
            } else {
                int b = m / SLEN, s = m - b * SLEN;
                int h = col >> 5, d = col & 31;
                if (mode == GM_CAT)
                    outB[(((long)(b * NHEAD + h) * SLEN + s) * 96) + catOff + d] = (bf16_t)v;
                else  // V transposed: [B,NH,32,SP]; pad cols hit only 0-probs
                    outB[((long)(b * NHEAD + h) * 32 + d) * SP + s] = (bf16_t)v;
            }
        }
}

// ---------------------------------------------------------------------------
// Per-row LayerNorm over 64 channels.
// ---------------------------------------------------------------------------
__global__ void ln_kernel(const float* __restrict__ in,
                          const float* __restrict__ g, const float* __restrict__ b,
                          float eps, float* __restrict__ outF,
                          bf16_t* __restrict__ outB, int M) {
    int m = blockIdx.x * 256 + threadIdx.x;
    if (m >= M) return;
    const float* x = in + (long)m * 64;
    float buf[64];
    float s = 0.f;
#pragma unroll
    for (int c = 0; c < 64; c += 4) {
        f32x4 v4 = *(const f32x4*)(x + c);
#pragma unroll
        for (int j = 0; j < 4; ++j) { buf[c + j] = v4[j]; s += v4[j]; }
    }
    float mean = s * (1.0f / 64.0f), v = 0.f;
    for (int c = 0; c < 64; ++c) { float d = buf[c] - mean; v += d * d; }
    float inv = rsqrtf(v * (1.0f / 64.0f) + eps);
    for (int c = 0; c < 64; ++c) {
        float y = (buf[c] - mean) * inv * g[c] + b[c];
        if (outF) outF[(long)m * 64 + c] = y;
        if (outB) outB[(long)m * 64 + c] = (bf16_t)y;
    }
}

// ---------------------------------------------------------------------------
// Broadcast pe projections into the 96-channel concat buffers (part [32..63]).
// ---------------------------------------------------------------------------
__global__ void scatter_pe(const bf16_t* __restrict__ pq,
                           const bf16_t* __restrict__ pk,
                           bf16_t* __restrict__ Qcat, bf16_t* __restrict__ Kcat) {
    int i = blockIdx.x * 256 + threadIdx.x;
    if (i >= BS * 64) return;
    int c = i & 63, m = i >> 6;
    int b = m / SLEN, s = m - b * SLEN;
    int h = c >> 5, d = c & 31;
    long dst = (((long)(b * NHEAD + h) * SLEN + s) * 96) + 32 + d;
    Qcat[dst] = pq[s * 64 + c];
    Kcat[dst] = pk[s * 64 + c];
}

// ---------------------------------------------------------------------------
// Attention: one wave per (b, head, 16-query tile).
// K slab (200x96) and transposed-V slab (32xSP) are DMA'd to LDS by the TDM;
// scores = Qcat[16x96].K^T -> causal softmax (all 32 lanes, shfl_xor pair
// reduce) -> probs(bf16, LDS) @ V.
// ---------------------------------------------------------------------------
__global__ __launch_bounds__(32) void attn_kernel(
    const bf16_t* __restrict__ Qcat, const bf16_t* __restrict__ Kcat,
    const bf16_t* __restrict__ Vt, bf16_t* __restrict__ ctx) {
    extern __shared__ char smem[];
    bf16_t* Kl  = (bf16_t*)(smem + LDS_K_OFF);    // [200][96]
    bf16_t* Vl  = (bf16_t*)(smem + LDS_V_OFF);    // [32][SP]
    float*  sc  = (float*) (smem + LDS_SC_OFF);   // [16][SP]
    bf16_t* scP = (bf16_t*)(smem + LDS_SP_OFF);   // [16][SPP]

    int lane = threadIdx.x, lm = lane & 15, half = lane >> 4;
    int qt = blockIdx.x % NQT;
    int bh = blockIdx.x / NQT;            // b*NH + h
    int h = bh & (NHEAD - 1), b = bh / NHEAD;

    const bf16_t* Qb = Qcat + (long)bh * SLEN * 96;
    const bf16_t* Kg = Kcat + (long)bh * SLEN * 96;
    const bf16_t* Vg = Vt   + (long)bh * 32 * SP;

#if HAVE_TDM
    tdm_load_1d((unsigned long long)(uintptr_t)Kg, LDS_K_OFF, SLEN * 96);
    tdm_load_1d((unsigned long long)(uintptr_t)Vg, LDS_V_OFF, 32 * SP);
#endif

    int qRow = qt * 16 + lm;
    v16bf qa[3];
    if (qRow < SLEN) {
#pragma unroll
        for (int c = 0; c < 3; ++c) qa[c] = frag_row(Qb + (long)qRow * 96, c * 32, half);
    } else {
#pragma unroll
        for (int c = 0; c < 3; ++c) qa[c] = frag_zero();
    }

#if HAVE_TDM
    __builtin_amdgcn_s_wait_tensorcnt(0);
#else
    for (int i = lane; i < SLEN * 96 / 8; i += 32)
        ((bf16x8*)Kl)[i] = ((const bf16x8*)Kg)[i];
    for (int i = lane; i < 32 * SP / 8; i += 32)
        ((bf16x8*)Vl)[i] = ((const bf16x8*)Vg)[i];
    __syncthreads();
#endif

    for (int st = 0; st < NQT; ++st) {
        int kRow = st * 16 + lm;
        v16bf kb[3];
        if (kRow < SLEN) {
#pragma unroll
            for (int c = 0; c < 3; ++c) kb[c] = frag_row(Kl + (long)kRow * 96, c * 32, half);
        } else {
#pragma unroll
            for (int c = 0; c < 3; ++c) kb[c] = frag_zero();
        }
        v8f c8 = {};
        c8 = wmma_bf16(qa[0], kb[0], c8);
        c8 = wmma_bf16(qa[1], kb[1], c8);
        c8 = wmma_bf16(qa[2], kb[2], c8);
#pragma unroll
        for (int r = 0; r < 8; ++r)
            sc[(r + half * 8) * SP + st * 16 + lm] = c8[r];
    }
    __syncthreads();

    {   // row softmax with causal mask; lane pair (lm, lm+16) shares row lm
        int q = qt * 16 + lm;
        float* row = sc + lm * SP;
        bf16_t* rowP = scP + lm * SPP;
        int lim = (q < SLEN) ? q : (SLEN - 1);     // inclusive
        const float scale = 0.1767766952966369f;   // 1/sqrt(32)
        float mx = -1e30f;
        for (int k = half; k <= lim; k += 2) {
            float v = row[k] * scale; row[k] = v; mx = fmaxf(mx, v);
        }
        mx = fmaxf(mx, __shfl_xor(mx, 16, 32));
        float sum = 0.f;
        for (int k = half; k <= lim; k += 2) {
            float e = __expf(row[k] - mx); row[k] = e; sum += e;
        }
        sum += __shfl_xor(sum, 16, 32);
        float inv = 1.0f / sum;
        for (int k = half; k <= lim; k += 2) rowP[k] = (bf16_t)(row[k] * inv);
        int k0 = lim + 1; if ((k0 & 1) != half) ++k0;
        for (int k = k0; k < SPP; k += 2) rowP[k] = (bf16_t)0.0f;
    }
    __syncthreads();

    v8f o[2] = {};
    for (int kc = 0; kc < 7; ++kc) {      // 7*32 = SPP
        v16bf pa = frag_row(scP + lm * SPP, kc * 32, half);
#pragma unroll
        for (int nt = 0; nt < 2; ++nt) {
            v16bf vb = frag_row(Vl + (nt * 16 + lm) * SP, kc * 32, half);
            o[nt] = wmma_bf16(pa, vb, o[nt]);
        }
    }
#pragma unroll
    for (int nt = 0; nt < 2; ++nt)
#pragma unroll
        for (int r = 0; r < 8; ++r) {
            int m = qt * 16 + r + half * 8;
            if (m < SLEN)
                ctx[((long)b * SLEN + m) * 64 + h * 32 + nt * 16 + lm] = (bf16_t)o[nt][r];
        }
}

// ---------------------------------------------------------------------------
// Final: log_feats = LN(q); pos/neg logits = dot with gathered item rows.
// ---------------------------------------------------------------------------
__global__ void logits_kernel(const float* __restrict__ q,
                              const int* __restrict__ posI, const int* __restrict__ negI,
                              const float* __restrict__ item_emb,
                              const float* __restrict__ g, const float* __restrict__ b,
                              float* __restrict__ out) {
    int m = blockIdx.x * 256 + threadIdx.x;
    if (m >= BS) return;
    const float* x = q + (long)m * 64;
    float buf[64];
    float s = 0.f;
#pragma unroll
    for (int c = 0; c < 64; c += 4) {
        f32x4 v4 = *(const f32x4*)(x + c);
#pragma unroll
        for (int j = 0; j < 4; ++j) { buf[c + j] = v4[j]; s += v4[j]; }
    }
    float mean = s * (1.0f / 64.0f), v = 0.f;
    for (int c = 0; c < 64; ++c) { float d = buf[c] - mean; v += d * d; }
    float inv = rsqrtf(v * (1.0f / 64.0f) + 1e-8f);
    const float* pe = item_emb + (long)posI[m] * 64;
    const float* ne = item_emb + (long)negI[m] * 64;
    float dp = 0.f, dn = 0.f;
    for (int c = 0; c < 64; ++c) {
        float y = (buf[c] - mean) * inv * g[c] + b[c];
        dp += y * pe[c]; dn += y * ne[c];
    }
    out[m] = dp;
    out[BS + m] = dn;
}

// ---------------------------------------------------------------------------
extern "C" void kernel_launch(void* const* d_in, const int* in_sizes, int n_in,
                              void* d_out, int out_size, void* d_ws, size_t ws_size,
                              hipStream_t stream) {
    (void)in_sizes; (void)n_in; (void)out_size; (void)ws_size;
    const int*   ts    = (const int*)d_in[0];
    const int*   posI  = (const int*)d_in[1];
    const int*   negI  = (const int*)d_in[2];
    const float* rev   = (const float*)d_in[3];
    const float* meta  = (const float*)d_in[4];
    const float* item  = (const float*)d_in[5];
    const float* peT   = (const float*)d_in[6];
    const float* featW = (const float*)d_in[7];
    const float* featb = (const float*)d_in[8];
    const float* revg  = (const float*)d_in[9];
    const float* revb  = (const float*)d_in[10];
    const float* Wm[10]; const float* Bm[10];
    for (int j = 0; j < 10; ++j) { Wm[j] = (const float*)d_in[11 + j]; Bm[j] = (const float*)d_in[21 + j]; }
    const float* attng = (const float*)d_in[31];
    const float* attnb = (const float*)d_in[32];
    const float* fwdg  = (const float*)d_in[33];
    const float* fwdb  = (const float*)d_in[34];
    const float* lastg = (const float*)d_in[35];
    const float* lastb = (const float*)d_in[36];

    char* ws = (char*)d_ws;
    size_t off = 0;
    auto alloc = [&](size_t bytes) -> void* {
        void* p = ws + off;
        off = (off + bytes + 255) & ~(size_t)255;
        return p;
    };
    bf16_t* WtAll = (bf16_t*)alloc((size_t)20 * 4096 * 2);
    bf16_t* fWt   = (bf16_t*)alloc((size_t)64 * EDIM * 2);
    bf16_t* featB = (bf16_t*)alloc((size_t)BS * 64 * 2);
    float*  qF    = (float*) alloc((size_t)BS * 64 * 4);
    bf16_t* qB    = (bf16_t*)alloc((size_t)BS * 64 * 2);
    float*  QnF   = (float*) alloc((size_t)BS * 64 * 4);
    bf16_t* QnB   = (bf16_t*)alloc((size_t)BS * 64 * 2);
    bf16_t* Qcat  = (bf16_t*)alloc((size_t)BDIM * NHEAD * SLEN * 96 * 2);
    bf16_t* Kcat  = (bf16_t*)alloc((size_t)BDIM * NHEAD * SLEN * 96 * 2);
    bf16_t* Vtb   = (bf16_t*)alloc((size_t)BDIM * NHEAD * 32 * SP * 2);
    bf16_t* ctxB  = (bf16_t*)alloc((size_t)BS * 64 * 2);
    float*  q2F   = (float*) alloc((size_t)BS * 64 * 4);
    float*  tF    = (float*) alloc((size_t)BS * 64 * 4);
    bf16_t* tB    = (bf16_t*)alloc((size_t)BS * 64 * 2);
    bf16_t* rB    = (bf16_t*)alloc((size_t)BS * 64 * 2);
    bf16_t* peB   = (bf16_t*)alloc((size_t)SLEN * 64 * 2);
    bf16_t* pqB   = (bf16_t*)alloc((size_t)SLEN * 64 * 2);
    bf16_t* pkB   = (bf16_t*)alloc((size_t)SLEN * 64 * 2);

    int gElem = (BS * 64 + 255) / 256;
    int gRow  = (BS + 255) / 256;
    int gTile = BS / 64;           // 800
    int gPrep = (20 * 4096 + 64 * EDIM + 255) / 256;

    prep_weights<<<gPrep, 256, 0, stream>>>(Wm[0], Wm[1], Wm[2], Wm[3], Wm[4],
                                            Wm[5], Wm[6], Wm[7], Wm[8], Wm[9],
                                            featW, WtAll, fWt);
    init_kernel<<<gElem, 256, 0, stream>>>(ts, item, peT, qF, qB, peB);
    feat_kernel<<<gTile, 128, FEAT_LDS_BYTES, stream>>>(
        ts, rev, meta, fWt, featb, revg, revb, featB);

    for (int i = 0; i < 2; ++i) {
        const bf16_t* Wt_q  = WtAll + (0 * 2 + i) * 4096;
        const bf16_t* Wt_k  = WtAll + (1 * 2 + i) * 4096;
        const bf16_t* Wt_v  = WtAll + (2 * 2 + i) * 4096;
        const bf16_t* Wt_pq = WtAll + (3 * 2 + i) * 4096;
        const bf16_t* Wt_pk = WtAll + (4 * 2 + i) * 4096;
        const bf16_t* Wt_fq = WtAll + (5 * 2 + i) * 4096;
        const bf16_t* Wt_fk = WtAll + (6 * 2 + i) * 4096;
        const bf16_t* Wt_o  = WtAll + (7 * 2 + i) * 4096;
        const bf16_t* Wt_1  = WtAll + (8 * 2 + i) * 4096;
        const bf16_t* Wt_2  = WtAll + (9 * 2 + i) * 4096;
        const float* bq = Bm[0] + i * 64;  const float* bk = Bm[1] + i * 64;
        const float* bv = Bm[2] + i * 64;  const float* bpq = Bm[3] + i * 64;
        const float* bpk = Bm[4] + i * 64; const float* bfq = Bm[5] + i * 64;
        const float* bfk = Bm[6] + i * 64; const float* bo = Bm[7] + i * 64;
        const float* b1 = Bm[8] + i * 64;  const float* b2 = Bm[9] + i * 64;

        ln_kernel<<<gRow, 256, 0, stream>>>(qF, attng + i * 64, attnb + i * 64,
                                            1e-8f, QnF, QnB, BS);
        gemm64_kernel<<<gTile, 128, 8192, stream>>>(QnB, Wt_q, bq, nullptr, nullptr,
                                                    nullptr, Qcat, BS, 0, GM_CAT, 0);
        gemm64_kernel<<<gTile, 128, 8192, stream>>>(qB, Wt_k, bk, nullptr, nullptr,
                                                    nullptr, Kcat, BS, 0, GM_CAT, 0);
        gemm64_kernel<<<gTile, 128, 8192, stream>>>(qB, Wt_v, bv, nullptr, nullptr,
                                                    nullptr, Vtb, BS, 0, GM_VT, 0);
        gemm64_kernel<<<gTile, 128, 8192, stream>>>(featB, Wt_fq, bfq, nullptr, nullptr,
                                                    nullptr, Qcat, BS, 0, GM_CAT, 64);
        gemm64_kernel<<<gTile, 128, 8192, stream>>>(featB, Wt_fk, bfk, nullptr, nullptr,
                                                    nullptr, Kcat, BS, 0, GM_CAT, 64);
        gemm64_kernel<<<4, 128, 8192, stream>>>(peB, Wt_pq, bpq, nullptr, nullptr,
                                                nullptr, pqB, SLEN, 0, GM_PLAIN, 0);
        gemm64_kernel<<<4, 128, 8192, stream>>>(peB, Wt_pk, bpk, nullptr, nullptr,
                                                nullptr, pkB, SLEN, 0, GM_PLAIN, 0);
        scatter_pe<<<gElem, 256, 0, stream>>>(pqB, pkB, Qcat, Kcat);

        attn_kernel<<<BDIM * NHEAD * NQT, 32, ATTN_LDS_BYTES, stream>>>(
            Qcat, Kcat, Vtb, ctxB);

        gemm64_kernel<<<gTile, 128, 8192, stream>>>(ctxB, Wt_o, bo, QnF, nullptr,
                                                    q2F, nullptr, BS, 0, GM_PLAIN, 0);
        ln_kernel<<<gRow, 256, 0, stream>>>(q2F, fwdg + i * 64, fwdb + i * 64,
                                            1e-8f, tF, tB, BS);
        gemm64_kernel<<<gTile, 128, 8192, stream>>>(tB, Wt_1, b1, nullptr, nullptr,
                                                    nullptr, rB, BS, 1, GM_PLAIN, 0);
        gemm64_kernel<<<gTile, 128, 8192, stream>>>(rB, Wt_2, b2, tF, ts,
                                                    qF, qB, BS, 0, GM_PLAIN, 0);
    }
    logits_kernel<<<gRow, 256, 0, stream>>>(qF, posI, negI, item, lastg, lastb,
                                            (float*)d_out);
}
